// GIN_Network_withEdgeFeatures_63402307224027
// MI455X (gfx1250) — matrix operand order repaired and checked by reading, not compiled, not run on hardware
//
#include <hip/hip_runtime.h>

// ---------------- problem constants ----------------
#define NN   50000
#define EE   800000
#define HH   64
#define EAC  16
#define LLAY 5
#define GG   512
#define IN2P 96      // 80 (=64+16) padded to 96 for 3 x K=32 WMMA steps
#define LH   320     // L*H

typedef _Float16 h16;
typedef __attribute__((ext_vector_type(2)))  _Float16 v2h;
typedef __attribute__((ext_vector_type(8)))  _Float16 v8h;
typedef __attribute__((ext_vector_type(16))) _Float16 v16h;
typedef __attribute__((ext_vector_type(8)))  float    v8f;

union V16H { v16h v; v8h h[2]; };

// A-fragment (16x32 f16) from an LDS tile stored row-major [16][stride].
// Layout: lane<16 -> row=lane, K = kbase+{0..15}
//         lane>=16 -> row=lane-16, K = kbase+{16..31}
__device__ inline v16h load_a_frag(const h16* tile, int lane, int stride, int kbase) {
    int row = lane & 15;
    int off = (lane < 16) ? 0 : 16;
    const h16* p = tile + row * stride + kbase + off;
    V16H u;
    u.h[0] = *(const v8h*)(p);
    u.h[1] = *(const v8h*)(p + 8);
    return u.v;
}

// B-fragment (32x16 f16) from LDS weights stored TRANSPOSED: [64 cols][stride K]
// lane<16 -> N=nt*16+lane, K = kbase+{0..15}; lane>=16 -> N=nt*16+lane-16, K = kbase+{16..31}
__device__ inline v16h load_b_frag(const h16* wT, int lane, int stride, int kbase, int nt) {
    int n    = nt * 16 + (lane & 15);
    int koff = (lane < 16) ? 0 : 16;
    return *(const v16h*)(wT + n * stride + kbase + koff);
}

#define WMMA_F16(A, B, C) \
    __builtin_amdgcn_wmma_f32_16x16x32_f16(false, (A), false, (B), (short)0, (C), false, false)

// ---------------- weight conversion: fp32 -> f16, transposed, K-padded ----------------
__global__ void convert_weights_kernel(
    const float* __restrict__ nn2w1, const float* __restrict__ nn2w2,
    const float* __restrict__ nn1w1, const float* __restrict__ nn1w2,
    h16* __restrict__ w2t1, h16* __restrict__ w2t2,
    h16* __restrict__ w1t1, h16* __restrict__ w1t2)
{
    int i = blockIdx.x * blockDim.x + threadIdx.x;
    const int R1 = LLAY * 64 * IN2P;   // 30720
    const int R2 = LLAY * 64 * 64;     // 20480
    if (i < R1) {
        int l = i / (64 * IN2P); int rem = i % (64 * IN2P);
        int n = rem / IN2P;      int k   = rem % IN2P;
        w2t1[i] = (k < (HH + EAC)) ? (h16)nn2w1[((size_t)l * 80 + k) * 64 + n] : (h16)0.0f;
        return;
    }
    i -= R1;
    if (i < R2) {
        int l = i / 4096; int rem = i % 4096; int n = rem / 64; int k = rem % 64;
        w2t2[i] = (h16)nn2w2[((size_t)l * 64 + k) * 64 + n];
        return;
    }
    i -= R2;
    if (i < R2) {
        int l = i / 4096; int rem = i % 4096; int n = rem / 64; int k = rem % 64;
        w1t1[i] = (h16)nn1w1[((size_t)l * 64 + k) * 64 + n];
        return;
    }
    i -= R2;
    if (i < R2) {
        int l = i / 4096; int rem = i % 4096; int n = rem / 64; int k = rem % 64;
        w1t2[i] = (h16)nn1w2[((size_t)l * 64 + k) * 64 + n];
    }
}

// ---------------- edge message kernel: 4 waves/block, 1 x 16-edge tile per wave ----------------
__global__ __launch_bounds__(128) void edge_msg_kernel(
    const float* __restrict__ x, const int* __restrict__ ei,
    const float* __restrict__ ea,
    const h16* __restrict__ w1T, const float* __restrict__ b1,
    const h16* __restrict__ w2T, const float* __restrict__ b2,
    float* __restrict__ agg)
{
    __shared__ alignas(32) h16 sW1[64 * IN2P];     // w1^T (K padded to 96)
    __shared__ alignas(32) h16 sW2[64 * 64];       // w2^T
    __shared__ alignas(32) h16 sA[4][16 * IN2P];   // per-wave input tile
    __shared__ alignas(32) h16 sH[4][16 * 64];     // per-wave hidden tile
    __shared__ int sSrc[4][16];
    __shared__ int sDst[4][16];

    const int tid  = threadIdx.x;
    const int w    = tid >> 5;
    const int lane = tid & 31;

    // stage weights (12KB + 8KB) cooperatively
    {
        const uint4* g1 = (const uint4*)w1T;  uint4* l1 = (uint4*)sW1;
        for (int i = tid; i < (64 * IN2P * 2) / 16; i += 128) l1[i] = g1[i];
        const uint4* g2 = (const uint4*)w2T;  uint4* l2 = (uint4*)sW2;
        for (int i = tid; i < (64 * 64 * 2) / 16; i += 128) l2[i] = g2[i];
    }
    __syncthreads();

    const int tile = blockIdx.x * 4 + w;   // 12500 blocks * 4 = 50000 tiles exactly
    const int e0   = tile * 16;

    if (lane < 16) {
        sSrc[w][lane] = ei[e0 + lane];
        sDst[w][lane] = ei[EE + e0 + lane];
    }

    // stage A tile: [16 edges][96] = [x[src] (64) | edge_attr (16) | zero pad (16)]
    for (int r = 0; r < 16; ++r) {
        int s = sSrc[w][r];
        float2 xv = *(const float2*)(x + (size_t)s * HH + lane * 2);
        *(v2h*)&sA[w][r * IN2P + lane * 2] = (v2h){(h16)xv.x, (h16)xv.y};
        if (lane < 16) {
            sA[w][r * IN2P + 64 + lane] = (h16)ea[(size_t)(e0 + r) * EAC + lane];
        } else {
            sA[w][r * IN2P + 80 + (lane - 16)] = (h16)0.0f;
        }
    }

    const int col0 = lane & 15;
    const int mb   = (lane < 16) ? 0 : 8;

    // GEMM1: [16x96] x [96x64] + b1, ReLU -> sH (f16)
    v16h a0 = load_a_frag(sA[w], lane, IN2P, 0);
    v16h a1 = load_a_frag(sA[w], lane, IN2P, 32);
    v16h a2 = load_a_frag(sA[w], lane, IN2P, 64);
    for (int nt = 0; nt < 4; ++nt) {
        v8f acc = {};
        acc = WMMA_F16(a0, load_b_frag(sW1, lane, IN2P, 0,  nt), acc);
        acc = WMMA_F16(a1, load_b_frag(sW1, lane, IN2P, 32, nt), acc);
        acc = WMMA_F16(a2, load_b_frag(sW1, lane, IN2P, 64, nt), acc);
        int col = nt * 16 + col0;
        float bb = b1[col];
        for (int r = 0; r < 8; ++r) {
            float vv = acc[r] + bb;
            vv = vv > 0.0f ? vv : 0.0f;
            sH[w][(mb + r) * 64 + col] = (h16)vv;
        }
    }

    // GEMM2: [16x64] x [64x64] + b2, mask self-loops, scatter-add to agg
    v16h h0 = load_a_frag(sH[w], lane, 64, 0);
    v16h h1 = load_a_frag(sH[w], lane, 64, 32);
    int dsts[8], srcs[8];
    for (int r = 0; r < 8; ++r) { dsts[r] = sDst[w][mb + r]; srcs[r] = sSrc[w][mb + r]; }
    for (int nt = 0; nt < 4; ++nt) {
        v8f acc = {};
        acc = WMMA_F16(h0, load_b_frag(sW2, lane, 64, 0,  nt), acc);
        acc = WMMA_F16(h1, load_b_frag(sW2, lane, 64, 32, nt), acc);
        int col = nt * 16 + col0;
        float bb = b2[col];
        for (int r = 0; r < 8; ++r) {
            if (dsts[r] != srcs[r]) {
                atomicAdd(&agg[(size_t)dsts[r] * HH + col], acc[r] + bb);
            }
        }
    }
}

// ---------------- node update kernel: x' = MLP1((1+eps)*x + agg), pool by graph ----------------
__global__ __launch_bounds__(128) void node_update_kernel(
    const float* __restrict__ xin, const float* __restrict__ agg,
    const h16* __restrict__ w1T, const float* __restrict__ b1,
    const h16* __restrict__ w2T, const float* __restrict__ b2,
    const float* __restrict__ eps, const int* __restrict__ batch,
    float* __restrict__ xout, float* __restrict__ pooled, int layer)
{
    __shared__ alignas(32) h16 sW1[64 * 64];
    __shared__ alignas(32) h16 sW2[64 * 64];
    __shared__ alignas(32) h16 sA[4][16 * 64];
    __shared__ alignas(32) h16 sH[4][16 * 64];

    const int tid  = threadIdx.x;
    const int w    = tid >> 5;
    const int lane = tid & 31;

    {
        const uint4* g1 = (const uint4*)w1T;  uint4* l1 = (uint4*)sW1;
        for (int i = tid; i < 512; i += 128) l1[i] = g1[i];
        const uint4* g2 = (const uint4*)w2T;  uint4* l2 = (uint4*)sW2;
        for (int i = tid; i < 512; i += 128) l2[i] = g2[i];
    }
    __syncthreads();

    const int tile = blockIdx.x * 4 + w;
    if (tile >= NN / 16) return;   // wave-uniform guard; EXEC stays all-1 inside

    const float sc = 1.0f + eps[0];
    const int n0 = tile * 16;

    for (int r = 0; r < 16; ++r) {
        size_t base = (size_t)(n0 + r) * HH + lane * 2;
        float2 xv = *(const float2*)(xin + base);
        float2 av = *(const float2*)(agg + base);
        *(v2h*)&sA[w][r * 64 + lane * 2] =
            (v2h){(h16)(sc * xv.x + av.x), (h16)(sc * xv.y + av.y)};
    }

    const int col0 = lane & 15;
    const int mb   = (lane < 16) ? 0 : 8;

    v16h a0 = load_a_frag(sA[w], lane, 64, 0);
    v16h a1 = load_a_frag(sA[w], lane, 64, 32);
    for (int nt = 0; nt < 4; ++nt) {
        v8f acc = {};
        acc = WMMA_F16(a0, load_b_frag(sW1, lane, 64, 0,  nt), acc);
        acc = WMMA_F16(a1, load_b_frag(sW1, lane, 64, 32, nt), acc);
        int col = nt * 16 + col0;
        float bb = b1[col];
        for (int r = 0; r < 8; ++r) {
            float vv = acc[r] + bb;
            vv = vv > 0.0f ? vv : 0.0f;
            sH[w][(mb + r) * 64 + col] = (h16)vv;
        }
    }

    v16h h0 = load_a_frag(sH[w], lane, 64, 0);
    v16h h1 = load_a_frag(sH[w], lane, 64, 32);
    int bt[8];
    for (int r = 0; r < 8; ++r) bt[r] = batch[n0 + mb + r];
    for (int nt = 0; nt < 4; ++nt) {
        v8f acc = {};
        acc = WMMA_F16(h0, load_b_frag(sW2, lane, 64, 0,  nt), acc);
        acc = WMMA_F16(h1, load_b_frag(sW2, lane, 64, 32, nt), acc);
        int col = nt * 16 + col0;
        float bb = b2[col];
        for (int r = 0; r < 8; ++r) {
            float vv = acc[r] + bb;
            int node = n0 + mb + r;
            xout[(size_t)node * HH + col] = vv;
            atomicAdd(&pooled[(size_t)bt[r] * LH + layer * HH + col], vv);
        }
    }
}

// ---------------- final MLP (tiny: 512x320x320 -> negligible FLOPs) ----------------
__global__ void fin1_kernel(const float* __restrict__ pooled, const float* __restrict__ w1,
                            const float* __restrict__ b1, float* __restrict__ hidden)
{
    int i = blockIdx.x * blockDim.x + threadIdx.x;
    if (i >= GG * LH) return;
    int g = i / LH, j = i % LH;
    const float* row = pooled + (size_t)g * LH;
    float s = b1[j];
    for (int k = 0; k < LH; ++k) s += row[k] * w1[(size_t)k * LH + j];
    hidden[i] = fmaxf(s, 0.0f);
}

__global__ void fin2_kernel(const float* __restrict__ hidden, const float* __restrict__ w2,
                            const float* __restrict__ b2, float* __restrict__ out)
{
    int g = blockIdx.x * blockDim.x + threadIdx.x;
    if (g >= GG) return;
    const float* row = hidden + (size_t)g * LH;
    float s = b2[0];
    for (int k = 0; k < LH; ++k) s += row[k] * w2[k];
    out[g] = s;
}

// ---------------- host launcher ----------------
extern "C" void kernel_launch(void* const* d_in, const int* in_sizes, int n_in,
                              void* d_out, int out_size, void* d_ws, size_t ws_size,
                              hipStream_t stream)
{
    (void)in_sizes; (void)n_in; (void)out_size; (void)ws_size;

    const float* x      = (const float*)d_in[0];
    const int*   ei     = (const int*)  d_in[1];
    const float* ea     = (const float*)d_in[2];
    const int*   batch  = (const int*)  d_in[3];
    const float* nn2w1  = (const float*)d_in[4];
    const float* nn2b1  = (const float*)d_in[5];
    const float* nn2w2  = (const float*)d_in[6];
    const float* nn2b2  = (const float*)d_in[7];
    const float* nn1w1  = (const float*)d_in[8];
    const float* nn1b1  = (const float*)d_in[9];
    const float* nn1w2  = (const float*)d_in[10];
    const float* nn1b2  = (const float*)d_in[11];
    const float* finw1  = (const float*)d_in[12];
    const float* finb1  = (const float*)d_in[13];
    const float* finw2  = (const float*)d_in[14];
    const float* finb2  = (const float*)d_in[15];
    const float* eps    = (const float*)d_in[16];

    // workspace layout (~40 MB)
    char* ws = (char*)d_ws;
    size_t off = 0;
    auto carve = [&](size_t bytes) -> void* {
        void* p = ws + off;
        off = (off + bytes + 255) & ~(size_t)255;
        return p;
    };
    float* xA     = (float*)carve((size_t)NN * HH * 4);
    float* xB     = (float*)carve((size_t)NN * HH * 4);
    float* agg    = (float*)carve((size_t)NN * HH * 4);
    float* pooled = (float*)carve((size_t)GG * LH * 4);
    float* hidden = (float*)carve((size_t)GG * LH * 4);
    h16*   w2t1   = (h16*)carve((size_t)LLAY * 64 * IN2P * 2);
    h16*   w2t2   = (h16*)carve((size_t)LLAY * 64 * 64 * 2);
    h16*   w1t1   = (h16*)carve((size_t)LLAY * 64 * 64 * 2);
    h16*   w1t2   = (h16*)carve((size_t)LLAY * 64 * 64 * 2);

    // weights -> f16 transposed/padded
    {
        int total = LLAY * 64 * IN2P + 3 * LLAY * 64 * 64;   // 92160
        int blocks = (total + 255) / 256;
        convert_weights_kernel<<<blocks, 256, 0, stream>>>(
            nn2w1, nn2w2, nn1w1, nn1w2, w2t1, w2t2, w1t1, w1t2);
    }

    hipMemcpyAsync(xA, x, (size_t)NN * HH * 4, hipMemcpyDeviceToDevice, stream);
    hipMemsetAsync(pooled, 0, (size_t)GG * LH * 4, stream);

    float* xcur = xA;
    float* xnxt = xB;
    for (int l = 0; l < LLAY; ++l) {
        hipMemsetAsync(agg, 0, (size_t)NN * HH * 4, stream);
        edge_msg_kernel<<<EE / 64, 128, 0, stream>>>(        // 12500 blocks x 4 waves
            xcur, ei, ea,
            w2t1 + (size_t)l * 64 * IN2P, nn2b1 + (size_t)l * 64,
            w2t2 + (size_t)l * 64 * 64,   nn2b2 + (size_t)l * 64,
            agg);
        node_update_kernel<<<(NN / 16 + 3) / 4, 128, 0, stream>>>(   // 782 blocks
            xcur, agg,
            w1t1 + (size_t)l * 64 * 64, nn1b1 + (size_t)l * 64,
            w1t2 + (size_t)l * 64 * 64, nn1b2 + (size_t)l * 64,
            eps, batch, xnxt, pooled, l);
        float* t = xcur; xcur = xnxt; xnxt = t;
    }

    fin1_kernel<<<(GG * LH + 255) / 256, 256, 0, stream>>>(pooled, finw1, finb1, hidden);
    fin2_kernel<<<(GG + 255) / 256, 256, 0, stream>>>(hidden, finw2, finb2, (float*)d_out);
}
